// CapsuleLayer_33277406609626
// MI455X (gfx1250) — compile-verified, matrix operand order checked by reading, not run on hardware
//
#include <hip/hip_runtime.h>
#include <hip/hip_bf16.h>

// CapsuleLayer dynamic routing for MI455X (gfx1250, wave32).
//
// Stage 1: u_hat[b,i,c,d] = sum_e W[i,c,d,e] * in[b,i,e]
//   -> batched GEMM per i: [256 x 8] x [8 x 160], via v_wmma_f32_16x16x32_f16
//      (K padded 8->32 with zeros; stage is store-BW bound so wasted MACs are free).
//   u_hat (f32, 180 MB) lands in d_ws and stays resident in the 192 MB L2 for
//   the routing passes.
//
// Stage 2: all 3 routing iterations fused, one workgroup per batch element.
//   Logits b[1152,10] + softmax coeffs live in LDS (~94 KB of the 320 KB WGP pool),
//   so the only memory traffic is u_hat reads (L2 hits).

typedef __attribute__((ext_vector_type(16))) _Float16 v16h;
typedef __attribute__((ext_vector_type(8)))  float    v8f;

#define B_SZ      256
#define IN_CAPS   1152
#define IN_DIM    8
#define NUM_CAPS  10
#define DIM_VEC   16
#define CD        (NUM_CAPS * DIM_VEC)   // 160

// ---------------------------------------------------------------------------
// Stage 1: u_hat via WMMA.  grid = (1152, 4), block = 128 (4 waves).
// Wave w of block (i, my) computes M-tile (my*4 + w) of the [256 x 160] GEMM
// for in-capsule i: loads its 16x8 A tile once, loops the 10 capsule N-tiles.
// ---------------------------------------------------------------------------
__global__ __launch_bounds__(128) void
uhat_wmma_kernel(const float* __restrict__ in,   // [256,1152,8]
                 const float* __restrict__ W,    // [1152,10,16,8]
                 float* __restrict__ uhat)       // [256,1152,10,16]
{
    const int i     = blockIdx.x;                    // in-capsule index
    const int wave  = threadIdx.x >> 5;
    const int lane  = threadIdx.x & 31;
    const int mtile = blockIdx.y * 4 + wave;         // 0..15 (M tile of 16 batch rows)
    const int l16   = lane & 15;
    const bool hi   = lane >= 16;

    // A fragment (16x32 f16): lanes 0-15, halves 0..7 carry K=0..7 (real data);
    // every other slot is K-padding = 0.  Lanes 16-31 carry only K>=8 -> zero.
    v16h a = {};
    if (!hi) {
        const float* ap = in + ((size_t)(mtile * 16 + l16) * IN_CAPS + i) * IN_DIM;
        #pragma unroll
        for (int e = 0; e < IN_DIM; ++e) a[e] = (_Float16)ap[e];
    }

    #pragma unroll
    for (int c = 0; c < NUM_CAPS; ++c) {
        // B fragment (32x16 f16): lanes 0-15 halves 0..7 hold K=0..7, rest zero, N=lane.
        v16h bm = {};
        if (!hi) {
            const float* bp = W + (((size_t)i * NUM_CAPS + c) * DIM_VEC + l16) * IN_DIM;
            #pragma unroll
            for (int e = 0; e < IN_DIM; ++e) bm[e] = (_Float16)bp[e];
        }
        v8f acc = {};
        acc = __builtin_amdgcn_wmma_f32_16x16x32_f16(
                  /*neg_a=*/false, a, /*neg_b=*/false, bm,
                  /*c_mod=*/(short)0, acc, /*reuse_a=*/false, /*reuse_b=*/false);

        // D layout: VGPR r -> M = r + (hi?8:0), N = l16.  b = mtile*16 + M.
        #pragma unroll
        for (int r = 0; r < 8; ++r) {
            const int m = r + (hi ? 8 : 0);
            const size_t bb = (size_t)(mtile * 16 + m);
            uhat[((bb * IN_CAPS + i) * NUM_CAPS + c) * DIM_VEC + l16] = acc[r];
        }
    }
}

// ---------------------------------------------------------------------------
// Stage 2: fused dynamic routing. grid = 256 (one WG per batch), block = 640.
// ---------------------------------------------------------------------------
#define RBLK 640

__global__ __launch_bounds__(RBLK) void
routing_kernel(const float* __restrict__ uhat,   // [256,1152,10,16]
               const float* __restrict__ bias,   // [1152,10]
               float* __restrict__ out)          // [256,10,16]
{
    const int b   = blockIdx.x;
    const int tid = threadIdx.x;

    __shared__ float blog[IN_CAPS * NUM_CAPS];   // routing logits  (46 KB)
    __shared__ float cw  [IN_CAPS * NUM_CAPS];   // softmax coeffs  (46 KB)
    __shared__ float spart[4 * CD];              // per-j partial s
    __shared__ float ssum[CD];                   // reduced s
    __shared__ float v_sh[CD];                   // squashed v

    for (int k = tid; k < IN_CAPS * NUM_CAPS; k += RBLK) blog[k] = bias[k];
    __syncthreads();

    const int j  = tid / CD;        // 0..3 : i-stripe
    const int cd = tid % CD;
    const int c  = cd / DIM_VEC;
    const int d  = cd % DIM_VEC;
    const float* up = uhat + (size_t)b * IN_CAPS * CD;

    for (int r = 0; r < 3; ++r) {
        // --- Phase A: c[i,:] = softmax(blog[i,:]) over the 10 capsules ---
        for (int i = tid; i < IN_CAPS; i += RBLK) {
            float row[NUM_CAPS];
            float mx = -1e30f;
            #pragma unroll
            for (int q = 0; q < NUM_CAPS; ++q) {
                row[q] = blog[i * NUM_CAPS + q];
                mx = fmaxf(mx, row[q]);
            }
            float sum = 0.f;
            #pragma unroll
            for (int q = 0; q < NUM_CAPS; ++q) { row[q] = __expf(row[q] - mx); sum += row[q]; }
            const float inv = 1.f / sum;
            #pragma unroll
            for (int q = 0; q < NUM_CAPS; ++q) cw[i * NUM_CAPS + q] = row[q] * inv;
        }
        __syncthreads();

        // --- Phase B: s[c,d] = sum_i cw[i,c] * uhat[i,c,d]  (L2-resident reads,
        //     16 consecutive threads share (j,c) -> contiguous 64B lines) ---
        float acc = 0.f;
        for (int i = j; i < IN_CAPS; i += 4)
            acc += cw[i * NUM_CAPS + c] * up[((size_t)i * NUM_CAPS + c) * DIM_VEC + d];
        spart[j * CD + cd] = acc;
        __syncthreads();

        // --- Phase C: reduce stripes, squash ---
        if (tid < CD)
            ssum[tid] = spart[tid] + spart[CD + tid] + spart[2 * CD + tid] + spart[3 * CD + tid];
        __syncthreads();
        if (tid < CD) {
            float sq = 0.f;
            #pragma unroll
            for (int q = 0; q < DIM_VEC; ++q) { const float t = ssum[c * DIM_VEC + q]; sq += t * t; }
            const float sc = sq / (1.f + sq) / sqrtf(sq + 1e-7f);
            const float vv = sc * ssum[tid];
            v_sh[tid] = vv;
            if (r == 2) out[(size_t)b * CD + tid] = vv;   // final output = squash(s)
        }
        __syncthreads();

        // --- Phase D: agreement[i,c] = sum_d uhat[i,c,d]*v[c,d];
        //     blog = agreement + blog + bias.  d spans 16-aligned lanes of a
        //     wave32 -> shfl_xor tree reduction, unique writer at d==0. ---
        if (r < 2) {
            const float vcd = v_sh[cd];
            for (int i = j; i < IN_CAPS; i += 4) {
                float p = up[((size_t)i * NUM_CAPS + c) * DIM_VEC + d] * vcd;
                #pragma unroll
                for (int off = 8; off >= 1; off >>= 1) p += __shfl_xor(p, off, 16);
                if (d == 0)
                    blog[i * NUM_CAPS + c] = p + blog[i * NUM_CAPS + c] + bias[i * NUM_CAPS + c];
            }
            __syncthreads();
        }
    }
}

// ---------------------------------------------------------------------------
extern "C" void kernel_launch(void* const* d_in, const int* in_sizes, int n_in,
                              void* d_out, int out_size, void* d_ws, size_t ws_size,
                              hipStream_t stream) {
    const float* in   = (const float*)d_in[0];   // [256,1152,8]
    const float* W    = (const float*)d_in[1];   // [1152,10,16,8]
    const float* bias = (const float*)d_in[2];   // [1152,10]
    float*       out  = (float*)d_out;           // [256,10,16]
    float*       uhat = (float*)d_ws;            // needs 256*1152*160*4 = 188,743,680 B

    dim3 g1(IN_CAPS, 4);
    uhat_wmma_kernel<<<g1, 128, 0, stream>>>(in, W, uhat);
    routing_kernel<<<B_SZ, RBLK, 0, stream>>>(uhat, bias, out);
}
